// BigramLanguageModel_63574105915950
// MI455X (gfx1250) — matrix operand order repaired and checked by reading, not compiled, not run on hardware
//
#include <hip/hip_runtime.h>
#include <stdint.h>
#include <math.h>

typedef _Float16 half_t;
typedef __attribute__((ext_vector_type(16))) _Float16 v16h;
typedef __attribute__((ext_vector_type(8)))  _Float16 v8h;
typedef __attribute__((ext_vector_type(8)))  float    v8f;

#define TSEQ      32
#define NHEAD     4
#define NBLK      8
#define VOCAB     65
#define BTOT      4096
#define NCHUNK    4
#define CHB       (BTOT / NCHUNK)   /* 1024 sequences per chunk */
#define MC        (CHB * TSEQ)      /* 32768 rows per chunk     */
#define SCALE_ATT 0.17677669529663689f  /* 32^-0.5 (source bug preserved) */

/* ---- f16 weight pack, TRANSPOSED [N][K] so tiles are contiguous row
 * copies for global_load_async_to_lds_b128 (element offsets) ---- */
#define QKV_SZ   (768 * 256)
#define P_SZ     (256 * 256)
#define W1_SZ    (1024 * 256)
#define W2_SZ    (256 * 1024)
#define BLK_W_SZ (QKV_SZ + P_SZ + W1_SZ + W2_SZ)  /* 786432 */
#define OFF_QKV  0
#define OFF_P    (QKV_SZ)
#define OFF_W1   (QKV_SZ + P_SZ)
#define OFF_W2   (QKV_SZ + P_SZ + W1_SZ)
#define LM_OFF   (8 * BLK_W_SZ)
#define LM_SZ    (128 * 256)   /* padded to BN=128 rows, zero-filled */
#define W16_TOTAL (LM_OFF + LM_SZ)

__device__ __forceinline__ v8f wmma16(v16h a, v16h b, v8f c) {
  return __builtin_amdgcn_wmma_f32_16x16x32_f16(false, a, false, b, (short)0, c,
                                                false, false);
}

__device__ __forceinline__ v16h pack16(v8h lo, v8h hi) {
  v16h r;
#pragma unroll
  for (int i = 0; i < 8; ++i) { r[i] = lo[i]; r[i + 8] = hi[i]; }
  return r;
}

__device__ __forceinline__ v8f zero8() {
  v8f z;
#pragma unroll
  for (int i = 0; i < 8; ++i) z[i] = 0.0f;
  return z;
}

/* LDS byte address for async-to-LDS: low 32 bits of generic shared ptr
 * (ISA: flat->LDS truncates addr[31:0]). */
__device__ __forceinline__ unsigned lds_addr(const void* p) {
  return (unsigned)(uintptr_t)p;
}

__device__ __forceinline__ void async_b128(unsigned lds, const void* gptr) {
  asm volatile("global_load_async_to_lds_b128 %0, %1, off"
               :: "v"(lds), "v"((unsigned long long)(uintptr_t)gptr)
               : "memory");
}

/* ================= generic WMMA GEMM =================
 * out[M,N] = A[M,K] @ W^T  (Wt stored [Npad][K], f16)  (+bias)(+resid)(relu?)
 * BM=64 BN=128 BK=32, 256 threads = 8 waves, each wave a 32x32 tile.
 * Double-buffered LDS; B (and A when f16) staged via async copy (ASYNCcnt),
 * overlapping the next tile's DMA with the current tile's WMMAs.
 */
template <typename TA, typename TO>
__global__ __launch_bounds__(256) void gemm_wmma(
    const TA* __restrict__ A, const half_t* __restrict__ Wt,
    const float* __restrict__ bias, const float* __restrict__ resid,
    TO* __restrict__ out, int M, int K, int N, int relu) {
  constexpr bool ASYNC_A = (sizeof(TA) == 2);
  __shared__ half_t As[2][64 * 32];    /* [m][k] */
  __shared__ half_t Bs[2][128 * 32];   /* [n][k] */

  const int tid  = threadIdx.x;
  const int lane = tid & 31;
  const int wid  = tid >> 5;
  const int wm   = wid >> 2;   /* 0..1 */
  const int wn   = wid & 3;    /* 0..3 */
  const int lh   = lane >> 4;
  const int l15  = lane & 15;

  const int m0 = blockIdx.y * 64;
  const int n0 = blockIdx.x * 128;

  v8f acc[2][2];
#pragma unroll
  for (int i = 0; i < 2; ++i)
#pragma unroll
    for (int j = 0; j < 2; ++j) acc[i][j] = zero8();

  auto stage = [&](int k0, int buf) {
    /* B tile: 128 rows x 32 halves = 8KB -> 2 async b128 per thread */
#pragma unroll
    for (int p = 0; p < 2; ++p) {
      const int c  = tid + p * 256;
      const int bn = c >> 2, ko = (c & 3) * 8;
      async_b128(lds_addr(&Bs[buf][bn * 32 + ko]),
                 Wt + (size_t)(n0 + bn) * K + k0 + ko);
    }
    /* A tile: 64 rows x 32 halves */
    const int row = tid >> 2, ko = (tid & 3) * 8;
    const TA* src = A + (size_t)(m0 + row) * K + k0 + ko;
    if constexpr (ASYNC_A) {
      async_b128(lds_addr(&As[buf][row * 32 + ko]), src);
    } else {
      float4 f0 = *(const float4*)(src);
      float4 f1 = *(const float4*)(src + 4);
      v8h hv;
      hv[0] = (half_t)f0.x; hv[1] = (half_t)f0.y;
      hv[2] = (half_t)f0.z; hv[3] = (half_t)f0.w;
      hv[4] = (half_t)f1.x; hv[5] = (half_t)f1.y;
      hv[6] = (half_t)f1.z; hv[7] = (half_t)f1.w;
      *(v8h*)&As[buf][row * 32 + ko] = hv;
    }
  };

  stage(0, 0);
  const int nk = K >> 5;
  for (int it = 0; it < nk; ++it) {
    const int buf  = it & 1;
    const bool more = (it + 1) < nk;
    if (more) stage((it + 1) << 5, buf ^ 1);
    /* in-order ASYNCcnt: leave only the next tile's copies outstanding */
    if (more) {
      if constexpr (ASYNC_A) asm volatile("s_wait_asynccnt 3" ::: "memory");
      else                   asm volatile("s_wait_asynccnt 2" ::: "memory");
    } else {
      asm volatile("s_wait_asynccnt 0" ::: "memory");
    }
    __syncthreads();

    v16h af[2], bf[2];
#pragma unroll
    for (int tm = 0; tm < 2; ++tm) {
      const int ar = wm * 32 + tm * 16 + l15;
      af[tm] = pack16(*(const v8h*)&As[buf][ar * 32 + lh * 8],
                      *(const v8h*)&As[buf][ar * 32 + 16 + lh * 8]);
    }
#pragma unroll
    for (int tn = 0; tn < 2; ++tn) {
      const int bn = wn * 32 + tn * 16 + l15;
      bf[tn] = pack16(*(const v8h*)&Bs[buf][bn * 32 + lh * 8],
                      *(const v8h*)&Bs[buf][bn * 32 + 16 + lh * 8]);
    }
#pragma unroll
    for (int tm = 0; tm < 2; ++tm)
#pragma unroll
      for (int tn = 0; tn < 2; ++tn)
        acc[tm][tn] = wmma16(af[tm], bf[tn], acc[tm][tn]);
    __syncthreads();
  }

  /* ---- epilogue: bias / residual / relu ---- */
#pragma unroll
  for (int tm = 0; tm < 2; ++tm)
#pragma unroll
    for (int tn = 0; tn < 2; ++tn) {
      const int gcol = n0 + wn * 32 + tn * 16 + l15;
      if (gcol < N) {
        const float bsv = bias ? bias[gcol] : 0.0f;
#pragma unroll
        for (int r = 0; r < 8; ++r) {
          const int grow = m0 + wm * 32 + tm * 16 + r + 8 * lh;
          float v = acc[tm][tn][r] + bsv;
          if (resid) v += resid[(size_t)grow * N + gcol];
          if (relu) v = fmaxf(v, 0.0f);
          out[(size_t)grow * N + gcol] = (TO)v;
        }
      }
    }
}

/* ============ fused attention: one wave per (sequence, head) ============
 * kqv: f16 [MC, 768] = k|q|v, head-major 64-wide slices.
 * attn: f16 [MC, 256] head-concat output.
 * S = k @ q^T * SCALE (source ordering), causal mask, softmax, P @ v.
 */
__global__ __launch_bounds__(32) void attention_wmma(
    const half_t* __restrict__ kqv, half_t* __restrict__ attn) {
  __shared__ half_t vT[64 * 32];  /* [d][s] */
  __shared__ half_t P[32 * 32];   /* [t][s] */

  const int lane = threadIdx.x;
  const int seq  = blockIdx.x >> 2;
  const int head = blockIdx.x & 3;
  const size_t rb = (size_t)seq * TSEQ;
  const int kcol = head * 64;
  const int qcol = 256 + head * 64;
  const int vcol = 512 + head * 64;
  const int lh = lane >> 4, l15 = lane & 15;

  for (int idx = lane; idx < 2048; idx += 32) {
    const int s = idx >> 6, d = idx & 63;
    vT[d * 32 + s] = kqv[(rb + s) * 768 + vcol + d];
  }
  __syncthreads();

  v8f sc[2][2];
#pragma unroll
  for (int i = 0; i < 2; ++i)
#pragma unroll
    for (int j = 0; j < 2; ++j) sc[i][j] = zero8();

#pragma unroll
  for (int kk = 0; kk < 64; kk += 32) {
    v16h af[2], bf[2];
#pragma unroll
    for (int tm = 0; tm < 2; ++tm) {
      const half_t* p = kqv + (rb + tm * 16 + l15) * 768 + kcol + kk;
      af[tm] = pack16(*(const v8h*)(p + lh * 8), *(const v8h*)(p + 16 + lh * 8));
    }
#pragma unroll
    for (int tn = 0; tn < 2; ++tn) {
      const half_t* p = kqv + (rb + tn * 16 + l15) * 768 + qcol + kk + lh * 8;
      bf[tn] = pack16(*(const v8h*)p, *(const v8h*)(p + 16));
    }
#pragma unroll
    for (int tm = 0; tm < 2; ++tm)
#pragma unroll
      for (int tn = 0; tn < 2; ++tn)
        sc[tm][tn] = wmma16(af[tm], bf[tn], sc[tm][tn]);
  }

#pragma unroll
  for (int tm = 0; tm < 2; ++tm)
#pragma unroll
    for (int r = 0; r < 8; ++r) {
      const int t  = tm * 16 + r + 8 * lh;
      const int s0 = l15, s1 = 16 + l15;
      float v0 = sc[tm][0][r] * SCALE_ATT;
      float v1 = sc[tm][1][r] * SCALE_ATT;
      if (s0 > t) v0 = -3.0e38f;
      if (s1 > t) v1 = -3.0e38f;
      float mx = fmaxf(v0, v1);
#pragma unroll
      for (int m = 1; m < 16; m <<= 1) mx = fmaxf(mx, __shfl_xor(mx, m, 32));
      float e0 = __expf(v0 - mx), e1 = __expf(v1 - mx);
      float sm = e0 + e1;
#pragma unroll
      for (int m = 1; m < 16; m <<= 1) sm += __shfl_xor(sm, m, 32);
      const float inv = 1.0f / sm;
      P[t * 32 + s0] = (half_t)(e0 * inv);
      P[t * 32 + s1] = (half_t)(e1 * inv);
    }
  __syncthreads();

  v16h pf[2];
#pragma unroll
  for (int tm = 0; tm < 2; ++tm) {
    const int t = tm * 16 + l15;
    pf[tm] = pack16(*(const v8h*)&P[t * 32 + lh * 8],
                    *(const v8h*)&P[t * 32 + 16 + lh * 8]);
  }
  v8f oc[2][4];
#pragma unroll
  for (int tn = 0; tn < 4; ++tn) {
    const int d = tn * 16 + l15;
    v16h bv = pack16(*(const v8h*)&vT[d * 32 + lh * 8],
                     *(const v8h*)&vT[d * 32 + 16 + lh * 8]);
#pragma unroll
    for (int tm = 0; tm < 2; ++tm) oc[tm][tn] = wmma16(pf[tm], bv, zero8());
  }
#pragma unroll
  for (int tm = 0; tm < 2; ++tm)
#pragma unroll
    for (int tn = 0; tn < 4; ++tn) {
      const int d = tn * 16 + l15;
#pragma unroll
      for (int r = 0; r < 8; ++r) {
        const int t = tm * 16 + r + 8 * lh;
        attn[(rb + t) * 256 + head * 64 + d] = (half_t)oc[tm][tn][r];
      }
    }
}

/* ================= LayerNorm: one wave per 256-wide row ================= */
__global__ __launch_bounds__(256) void layernorm_k(
    const float* __restrict__ src, const float* __restrict__ g,
    const float* __restrict__ b, float* __restrict__ dst, int M) {
  const int wid = threadIdx.x >> 5, lane = threadIdx.x & 31;
  const int row = blockIdx.x * 8 + wid;
  if (row >= M) return;
  const float* x = src + (size_t)row * 256;
  float v[8], sum = 0.0f;
#pragma unroll
  for (int j = 0; j < 8; ++j) { v[j] = x[j * 32 + lane]; sum += v[j]; }
#pragma unroll
  for (int m = 1; m < 32; m <<= 1) sum += __shfl_xor(sum, m, 32);
  const float mean = sum * (1.0f / 256.0f);
  float var = 0.0f;
#pragma unroll
  for (int j = 0; j < 8; ++j) { float d = v[j] - mean; var += d * d; }
#pragma unroll
  for (int m = 1; m < 32; m <<= 1) var += __shfl_xor(var, m, 32);
  const float rstd = rsqrtf(var * (1.0f / 256.0f) + 1e-5f);
  float* o = dst + (size_t)row * 256;
#pragma unroll
  for (int j = 0; j < 8; ++j) {
    const int c = j * 32 + lane;
    o[c] = (v[j] - mean) * rstd * g[c] + b[c];
  }
}

/* ================= embedding (pos table indexed by TOKEN: bug kept) ===== */
__global__ void embed_k(const int* __restrict__ x, const float* __restrict__ tok,
                        const float* __restrict__ pos, float* __restrict__ h,
                        int row0) {
  const int total = MC * 256;
  for (int idx = blockIdx.x * blockDim.x + threadIdx.x; idx < total;
       idx += gridDim.x * blockDim.x) {
    const int r = idx >> 8, c = idx & 255;
    const int tid = x[row0 + r];
    h[idx] = tok[tid * 256 + c] + pos[tid * 256 + c];
  }
}

/* ================= loss: wave per row over 65 logits ================= */
__global__ __launch_bounds__(256) void loss_k(
    const float* __restrict__ logits, const int* __restrict__ y, int row0,
    float* __restrict__ acc) {
  const int wid = threadIdx.x >> 5, lane = threadIdx.x & 31;
  const int rr = blockIdx.x * 8 + wid;
  if (rr >= MC) return;
  const float* lg = logits + (size_t)rr * VOCAB;
  const float v0 = lg[lane];
  const float v1 = lg[lane + 32];
  const float v2 = (lane == 0) ? lg[64] : -3.0e38f;
  float mx = fmaxf(fmaxf(v0, v1), v2);
#pragma unroll
  for (int m = 1; m < 32; m <<= 1) mx = fmaxf(mx, __shfl_xor(mx, m, 32));
  float s = __expf(v0 - mx) + __expf(v1 - mx) +
            ((lane == 0) ? __expf(v2 - mx) : 0.0f);
#pragma unroll
  for (int m = 1; m < 32; m <<= 1) s += __shfl_xor(s, m, 32);
  if (lane == 0) {
    const float lse = mx + logf(s);
    atomicAdd(acc, lse - lg[y[row0 + rr]]);
  }
}

__global__ void zero_k(float* a) { *a = 0.0f; }
__global__ void finalize_k(const float* acc, float* out_loss) {
  *out_loss = *acc * (1.0f / (4096.0f * 32.0f));
}

/* ============ weight packing: f32 -> f16, transposed to [N][K] ============ */
__global__ void cvt_qkv_k(const float* __restrict__ Wk, const float* __restrict__ Wq,
                          const float* __restrict__ Wv, half_t* __restrict__ w16) {
  const int total = NBLK * 768 * 256;
  for (int idx = blockIdx.x * blockDim.x + threadIdx.x; idx < total;
       idx += gridDim.x * blockDim.x) {
    const int blk = idx / (768 * 256), rem = idx % (768 * 256);
    const int n = rem / 256, c = rem % 256;     /* dst row n, dst col k=c */
    const int sel = n >> 8, nn = n & 255;
    const int head = nn >> 6, d = nn & 63;
    const float* src = (sel == 0) ? Wk : (sel == 1) ? Wq : Wv;
    const float v = src[(((size_t)blk * 4 + head) * 256 + c) * 64 + d];
    w16[(size_t)blk * BLK_W_SZ + OFF_QKV + rem] = (half_t)v;
  }
}

/* src [NBLK][K][N] -> dst [NBLK][N][K] */
__global__ void cvt_t_k(const float* __restrict__ src, half_t* __restrict__ w16,
                        int off, int K, int N) {
  const int per = K * N;
  const int total = NBLK * per;
  for (int idx = blockIdx.x * blockDim.x + threadIdx.x; idx < total;
       idx += gridDim.x * blockDim.x) {
    const int blk = idx / per, rem = idx % per;
    const int n = rem / K, k = rem % K;
    w16[(size_t)blk * BLK_W_SZ + off + rem] =
        (half_t)src[((size_t)blk * K + k) * N + n];
  }
}

/* lmW [256][65] -> [128][256] zero-padded rows */
__global__ void cvt_lm_k(const float* __restrict__ lmW, half_t* __restrict__ w16) {
  const int total = 128 * 256;
  for (int idx = blockIdx.x * blockDim.x + threadIdx.x; idx < total;
       idx += gridDim.x * blockDim.x) {
    const int n = idx / 256, k = idx % 256;
    w16[LM_OFF + idx] = (half_t)((n < VOCAB) ? lmW[k * VOCAB + n] : 0.0f);
  }
}

/* ================= host ================= */
extern "C" void kernel_launch(void* const* d_in, const int* in_sizes, int n_in,
                              void* d_out, int out_size, void* d_ws,
                              size_t ws_size, hipStream_t stream) {
  const int*   x    = (const int*)d_in[0];
  const int*   y    = (const int*)d_in[1];
  const float* tok  = (const float*)d_in[2];
  const float* pos  = (const float*)d_in[3];
  const float* Wk   = (const float*)d_in[4];
  const float* Wq   = (const float*)d_in[5];
  const float* Wv   = (const float*)d_in[6];
  const float* Wp   = (const float*)d_in[7];
  const float* bp   = (const float*)d_in[8];
  const float* ln1g = (const float*)d_in[9];
  const float* ln1b = (const float*)d_in[10];
  const float* W1   = (const float*)d_in[11];
  const float* b1   = (const float*)d_in[12];
  const float* W2   = (const float*)d_in[13];
  const float* b2   = (const float*)d_in[14];
  const float* ln2g = (const float*)d_in[15];
  const float* ln2b = (const float*)d_in[16];
  const float* lmW  = (const float*)d_in[17];
  const float* lmb  = (const float*)d_in[18];
  float* out = (float*)d_out;

  char* ws = (char*)d_ws;
  size_t off = 0;
  auto take = [&](size_t bytes) {
    size_t o = off;
    off = (off + bytes + 255) & ~(size_t)255;
    return o;
  };
  float*  lossacc = (float*)(ws + take(sizeof(float)));
  half_t* w16     = (half_t*)(ws + take((size_t)W16_TOTAL * sizeof(half_t)));
  float*  h       = (float*)(ws + take((size_t)MC * 256 * sizeof(float)));
  float*  res     = (float*)(ws + take((size_t)MC * 256 * sizeof(float)));
  half_t* kqv     = (half_t*)(ws + take((size_t)MC * 768 * sizeof(half_t)));
  half_t* ffa     = (half_t*)(ws + take((size_t)MC * 1024 * sizeof(half_t)));
  /* attn-concat f16 [MC,256] aliases the front of ffa (dead before ff1). */

  cvt_qkv_k<<<2048, 256, 0, stream>>>(Wk, Wq, Wv, w16);
  cvt_t_k<<<2048, 256, 0, stream>>>(Wp, w16, OFF_P, 256, 256);
  cvt_t_k<<<4096, 256, 0, stream>>>(W1, w16, OFF_W1, 256, 1024);
  cvt_t_k<<<4096, 256, 0, stream>>>(W2, w16, OFF_W2, 1024, 256);
  cvt_lm_k<<<128, 256, 0, stream>>>(lmW, w16);
  zero_k<<<1, 1, 0, stream>>>(lossacc);

  for (int ch = 0; ch < NCHUNK; ++ch) {
    const int row0 = ch * MC;
    embed_k<<<4096, 256, 0, stream>>>(x, tok, pos, h, row0);

    for (int b = 0; b < NBLK; ++b) {
      const half_t* wb = w16 + (size_t)b * BLK_W_SZ;
      /* fused QKV: [MC,256] @ [256,768] -> f16 */
      gemm_wmma<float, half_t><<<dim3(768 / 128, MC / 64), 256, 0, stream>>>(
          h, wb + OFF_QKV, nullptr, nullptr, kqv, MC, 256, 768, 0);
      attention_wmma<<<CHB * NHEAD, 32, 0, stream>>>(kqv, ffa);
      /* proj + bias + residual(h) -> res (f32) */
      gemm_wmma<half_t, float><<<dim3(256 / 128, MC / 64), 256, 0, stream>>>(
          ffa, wb + OFF_P, bp + b * 256, h, res, MC, 256, 256, 0);
      layernorm_k<<<MC / 8, 256, 0, stream>>>(res, ln1g + b * 256,
                                              ln1b + b * 256, h, MC);
      /* FFN up + ReLU -> f16 */
      gemm_wmma<float, half_t><<<dim3(1024 / 128, MC / 64), 256, 0, stream>>>(
          h, wb + OFF_W1, b1 + b * 1024, nullptr, ffa, MC, 256, 1024, 1);
      /* FFN down + bias + residual(h) -> res */
      gemm_wmma<half_t, float><<<dim3(256 / 128, MC / 64), 256, 0, stream>>>(
          ffa, wb + OFF_W2, b2 + b * 256, h, res, MC, 1024, 256, 0);
      layernorm_k<<<MC / 8, 256, 0, stream>>>(res, ln2g + b * 256,
                                              ln2b + b * 256, h, MC);
    }
    /* LM head: [MC,256] @ [256,65] (pack padded to 128 rows) -> logits */
    gemm_wmma<float, float><<<dim3(1, MC / 64), 256, 0, stream>>>(
        h, w16 + LM_OFF, lmb, nullptr, out + (size_t)row0 * VOCAB, MC, 256,
        VOCAB, 0);
    loss_k<<<MC / 8, 256, 0, stream>>>(out + (size_t)row0 * VOCAB, y, row0,
                                       lossacc);
  }
  finalize_k<<<1, 1, 0, stream>>>(lossacc, out + (out_size - 1));
}
